// LSTMLayer_37623913513463
// MI455X (gfx1250) — matrix operand order repaired and checked by reading, not compiled, not run on hardware
//
#include <hip/hip_runtime.h>
#include <hip/hip_bf16.h>

typedef __attribute__((ext_vector_type(16))) __bf16 v16bf;
typedef __attribute__((ext_vector_type(8)))  float  v8f;

#define T_STEPS 1024
#define BATCH   128
#define FEAT    256
#define LAYER   256
#define TOTK    512
#define NBLK    128   // 8 row tiles x 16 col tiles, one wave32 each
#define HQUAD   4096  // uint4s per h buffer parity (128*256 bf16 / 8)

// ---------------------------------------------------------------------------
// Prep kernel: transpose + convert weights.
// Wt[g][n][k] = (bf16) W_g[k][n],  g in {i,r,f,o}, n in [0,256), k in [0,512)
// ---------------------------------------------------------------------------
__global__ void cvt_w_kernel(const float* __restrict__ Wi, const float* __restrict__ Wr,
                             const float* __restrict__ Wf, const float* __restrict__ Wo,
                             __hip_bfloat16* __restrict__ Wt) {
  int id = blockIdx.x * blockDim.x + threadIdx.x;   // 0 .. 4*256*512-1
  if (id >= 4 * LAYER * TOTK) return;
  int g   = id / (LAYER * TOTK);
  int rem = id - g * (LAYER * TOTK);
  int n   = rem / TOTK;
  int k   = rem - n * TOTK;
  const float* W = (g == 0) ? Wi : (g == 1) ? Wr : (g == 2) ? Wf : Wo;
  Wt[id] = __float2bfloat16(W[k * LAYER + n]);
}

// ---------------------------------------------------------------------------
// Prep kernel: convert x_batch fp32 -> bf16 (one pass; stays L2 resident).
// ---------------------------------------------------------------------------
__global__ void cvt_x_kernel(const float* __restrict__ x,
                             __hip_bfloat16* __restrict__ xb, int n) {
  int i      = blockIdx.x * blockDim.x + threadIdx.x;
  int stride = gridDim.x * blockDim.x;
  for (; i < n; i += stride) xb[i] = __float2bfloat16(x[i]);
}

// ---------------------------------------------------------------------------
// Prep kernel: zero sync counters + both h parities (every launch, so graph
// replay is deterministic).
// ---------------------------------------------------------------------------
__global__ void init_ws_kernel(unsigned* __restrict__ cnt, unsigned* __restrict__ gen,
                               unsigned* __restrict__ hz, int nwords) {
  int i      = blockIdx.x * blockDim.x + threadIdx.x;
  int stride = gridDim.x * blockDim.x;
  if (i == 0) { *cnt = 0u; *gen = 0u; }
  for (; i < nwords; i += stride) hz[i] = 0u;
}

// ---------------------------------------------------------------------------
// Persistent recurrent kernel: 128 single-wave workgroups, one 16x16 output
// tile (all 4 gates) per wave, weights in LDS, cell state in VGPRs,
// h double-buffered (index parity, single pointer) in global, one
// device-wide barrier per timestep.
// ---------------------------------------------------------------------------
__global__ __launch_bounds__(32) void lstm_persist_kernel(
    const uint4* __restrict__ xq,        // bf16 x [T][B][F], as uint4 (8 elem)
    const uint4* __restrict__ wq,        // bf16 Wt [4][256][512], as uint4
    __hip_bfloat16* __restrict__ hbuf,   // bf16 h, 2 parities of [B][L]
    const float* __restrict__ bias,      // b_i [256]
    unsigned* __restrict__ cnt, unsigned* __restrict__ gen,
    float* __restrict__ out)             // h_last [B][L] fp32
{
  __shared__ uint4 Bsh[4096];            // 64 KB: 4 gates x 16 cols x 512 K (bf16)

  const int lane = threadIdx.x;          // wave32
  const int blk  = blockIdx.x;
  const int rb   = blk >> 4;             // batch tile 0..7
  const int cb   = blk & 15;             // layer-col tile 0..15
  const int n    = lane & 15;            // local output column
  const int hi   = lane >> 4;            // half-wave selector
  const int rowA = rb * 16 + (lane & 15);// batch row this lane loads for A

  // Preload this tile's weight fragments into LDS (contiguous 16 KB per gate).
  for (int i = lane; i < 4096; i += 32)
    Bsh[i] = wq[(i >> 10) * 16384 + cb * 1024 + (i & 1023)];
  __syncthreads();

  const uint4* hq = (const uint4*)hbuf;  // single global base (SGPR saddr)
  const float bval = bias[cb * 16 + n];

  float c[8];
  #pragma unroll
  for (int r = 0; r < 8; ++r) c[r] = 0.f;

  for (int t = 0; t < T_STEPS; ++t) {
    const int rdq = (t & 1) * HQUAD;                  // read parity, uint4 units
    const int wre = ((t + 1) & 1) * (BATCH * LAYER);  // write parity, elements

    // Prefetch next timestep's x tile (16 rows x 512B, contiguous).
    if (t + 1 < T_STEPS) {
      const char* pf = (const char*)(xq + (size_t)((t + 1) * BATCH + rb * 16) * 32)
                       + lane * 256;
      __builtin_prefetch(pf, 0, 1);
    }

    v8f acc0 = {}, acc1 = {}, acc2 = {}, acc3 = {};

    const int xbase = (t * BATCH + rowA) * 32;        // uint4 units
    const int hrow  = rdq + rowA * 32;                // uint4 units

    #pragma unroll
    for (int kc = 0; kc < 16; ++kc) {
      // A fragment (bf16 16x32): lane m<16 holds K runs [0..7],[16..23];
      // lanes 16-31 hold [8..15],[24..31]  -> two 16B loads per lane.
      union { uint4 q[2]; v16bf v; } a;
      if (kc < 8) {                                   // x part of z
        a.q[0] = xq[xbase + kc * 4 + hi];
        a.q[1] = xq[xbase + kc * 4 + hi + 2];
      } else {                                        // h part of z
        const int hbase = hrow + (kc - 8) * 4 + hi;
        a.q[0] = hq[hbase];
        a.q[1] = hq[hbase + 2];
      }
      #pragma unroll
      for (int g = 0; g < 4; ++g) {
        // B fragment (bf16 32x16): lanes 0-15 K [0..7],[16..23] of col n, lanes 16-31
        // K [8..15],[24..31] -> two 16B loads in [N,K]-major LDS image.
        union { uint4 q[2]; v16bf v; } b;
        const int bidx = (g * 16 + n) * 64 + kc * 4 + hi;
        b.q[0] = Bsh[bidx];
        b.q[1] = Bsh[bidx + 2];
        v8f& acc = (g == 0) ? acc0 : (g == 1) ? acc1 : (g == 2) ? acc2 : acc3;
        acc = __builtin_amdgcn_wmma_f32_16x16x32_bf16(
                  false, a.v, false, b.v, (short)0, acc, false, false);
      }
    }

    // Elementwise cell update. C/D layout: VGPR r -> M=r (lanes 0-15) /
    // M=r+8 (lanes 16-31); N = lane&15.
    const int colG = cb * 16 + n;
    #pragma unroll
    for (int r = 0; r < 8; ++r) {
      const int rowG = rb * 16 + r + hi * 8;
      float pi = acc0[r] + bval;
      float iv = pi > 0.f ? pi : 0.f;                       // relu input unit
      float rv = 1.f / (1.f + __expf(-acc1[r]));            // remember gate
      float fv = 1.f / (1.f + __expf(-acc2[r]));            // forget gate
      float ov = 1.f / (1.f + __expf(-acc3[r]));            // output gate
      c[r]     = c[r] * fv + iv * rv;
      float hv = tanhf(c[r]) * ov;
      hbuf[wre + rowG * LAYER + colG] = __float2bfloat16(hv);
      if (t == T_STEPS - 1) out[rowG * LAYER + colG] = hv;
    }

    // Device-wide barrier: monotonic arrive counter + generation release.
    __threadfence();
    const unsigned tgt = (unsigned)(t + 1);
    if (lane == 0) {
      unsigned old = __hip_atomic_fetch_add(cnt, 1u, __ATOMIC_ACQ_REL,
                                            __HIP_MEMORY_SCOPE_AGENT);
      if (old == (unsigned)NBLK * tgt - 1u) {
        __hip_atomic_store(gen, tgt, __ATOMIC_RELEASE, __HIP_MEMORY_SCOPE_AGENT);
      }
      while (__hip_atomic_load(gen, __ATOMIC_ACQUIRE,
                               __HIP_MEMORY_SCOPE_AGENT) < tgt) {
        __builtin_amdgcn_s_sleep(2);
      }
    }
    __threadfence();
    __syncthreads();
  }
}

// ---------------------------------------------------------------------------
extern "C" void kernel_launch(void* const* d_in, const int* in_sizes, int n_in,
                              void* d_out, int out_size, void* d_ws, size_t ws_size,
                              hipStream_t stream) {
  const float* x  = (const float*)d_in[0];  // [T,B,F]
  const float* Wi = (const float*)d_in[1];  // [TOT,L]
  const float* bi = (const float*)d_in[2];  // [L]
  const float* Wr = (const float*)d_in[3];
  const float* Wf = (const float*)d_in[4];
  const float* Wo = (const float*)d_in[5];

  char* ws = (char*)d_ws;
  unsigned*       cnt  = (unsigned*)ws;                         // arrive counter
  unsigned*       gen  = (unsigned*)(ws + 64);                  // generation
  __hip_bfloat16* Wt   = (__hip_bfloat16*)(ws + 256);           // 1 MiB
  __hip_bfloat16* hbuf = (__hip_bfloat16*)(ws + 256 + 1048576); // 128 KiB (2 parities)
  __hip_bfloat16* xb   = hbuf + 2 * BATCH * LAYER;              // 64 MiB

  // 1) zero counters + both h parities (each launch -> replay-deterministic)
  init_ws_kernel<<<64, 256, 0, stream>>>(cnt, gen, (unsigned*)hbuf,
                                         (2 * BATCH * LAYER * 2) / 4);

  // 2) weights: fp32 [TOT,L] -> bf16 [4][L][TOT] (transposed for B fragments)
  cvt_w_kernel<<<(4 * LAYER * TOTK + 255) / 256, 256, 0, stream>>>(Wi, Wr, Wf, Wo, Wt);

  // 3) x: fp32 -> bf16
  const int nx = T_STEPS * BATCH * FEAT;
  cvt_x_kernel<<<4096, 256, 0, stream>>>(x, xb, nx);

  // 4) persistent recurrent kernel
  lstm_persist_kernel<<<NBLK, 32, 0, stream>>>(
      (const uint4*)xb, (const uint4*)Wt, hbuf, bi, cnt, gen, (float*)d_out);
}
